// CRF_10333691314671
// MI455X (gfx1250) — matrix operand order repaired and checked by reading, not compiled, not run on hardware
//
#include <hip/hip_runtime.h>

typedef __attribute__((ext_vector_type(16))) _Float16 v16h;
typedef __attribute__((ext_vector_type(8)))  float    v8f;

#define NTAGS 128
#define SEQ   512
#define BATCH 256
#define MROWS 16

union HFrag { v16h v; _Float16 h[16]; };
union FAcc  { v8f  v; float    f[8];  };

// ---------------------------------------------------------------------------
// Forward algorithm in the exp domain via f16 WMMA.
// Each block owns 16 batch rows; 8 waves, one 16-column N-tile per wave.
// alpha (f32) and A=exp(alpha-rowmax) (f16) live in LDS; B fragments
// (exp(trans - colmax), f16) are loaded into VGPRs once and reused all steps.
// ---------------------------------------------------------------------------
__global__ __launch_bounds__(256) void crf_forward_kernel(
    const float* __restrict__ emissions,
    const float* __restrict__ transitions,
    float* __restrict__ partials)
{
    __shared__ _Float16 Eh[NTAGS * NTAGS];     // 32 KB  exp(trans - cmax[j])
    __shared__ float    cmax[NTAGS];           // column max of transitions
    __shared__ float    alpha[MROWS * NTAGS];  // 8 KB   running alpha (f32)
    __shared__ _Float16 Ah[MROWS * NTAGS];     // 4 KB   exp(alpha - rowmax)
    __shared__ float    rowm[MROWS];
    __shared__ float    lsered[MROWS];

    const int tid   = threadIdx.x;
    const int lane  = tid & 31;
    const int wv    = tid >> 5;        // wave id 0..7 == N-tile
    const int hi    = lane >> 4;       // half-wave
    const int l15   = lane & 15;
    const int bRow0 = blockIdx.x * MROWS;

    // --- column max of transitions ---
    if (tid < NTAGS) {
        float m = -3.0e38f;
        for (int i = 0; i < NTAGS; ++i) m = fmaxf(m, transitions[i * NTAGS + tid]);
        cmax[tid] = m;
    }
    __syncthreads();

    // --- Eh = exp(trans - cmax[col]) in f16 ---
    for (int idx = tid; idx < NTAGS * NTAGS; idx += 256)
        Eh[idx] = (_Float16)__expf(transitions[idx] - cmax[idx & (NTAGS - 1)]);

    // --- alpha0 = emissions[:, 0, :] ---
    for (int k = tid; k < MROWS * NTAGS; k += 256) {
        int r = k >> 7, c = k & (NTAGS - 1);
        alpha[k] = emissions[(size_t)(bRow0 + r) * SEQ * NTAGS + c];
    }
    __syncthreads();

    // --- preload loop-invariant B fragments (32x16 f16 per K-tile) ---
    // B layout: lane = N (l15); halves e -> K = kt*32 + hi*16 + e
    HFrag bfrag[4];
    #pragma unroll
    for (int kt = 0; kt < 4; ++kt) {
        #pragma unroll
        for (int e = 0; e < 16; ++e) {
            int K = kt * 32 + hi * 16 + e;
            bfrag[kt].h[e] = Eh[K * NTAGS + wv * 16 + l15];
        }
    }

    const int myrow = tid >> 4;          // 16 threads per batch row
    const int cbase = (tid & 15) * 8;    // 8 columns each
    const int Nglob = wv * 16 + l15;

    for (int t = 1; t < SEQ; ++t) {
        __builtin_prefetch(
            &emissions[((size_t)(bRow0 + myrow) * SEQ + t) * NTAGS + cbase], 0, 1);

        // phase 1: per-row max + A = exp(alpha - m) into LDS (f16)
        float av[8], m = -3.0e38f;
        #pragma unroll
        for (int i = 0; i < 8; ++i) {
            av[i] = alpha[myrow * NTAGS + cbase + i];
            m = fmaxf(m, av[i]);
        }
        #pragma unroll
        for (int off = 8; off >= 1; off >>= 1) m = fmaxf(m, __shfl_xor(m, off, 16));
        if ((tid & 15) == 0) rowm[myrow] = m;
        #pragma unroll
        for (int i = 0; i < 8; ++i)
            Ah[myrow * NTAGS + cbase + i] = (_Float16)__expf(av[i] - m);
        __syncthreads();

        // phase 2: load A fragments (16x32 f16, K = kt*32 + hi*16 + e) + 4x WMMA
        HFrag afrag[4];
        #pragma unroll
        for (int kt = 0; kt < 4; ++kt) {
            #pragma unroll
            for (int e = 0; e < 16; ++e) {
                int K = kt * 32 + hi * 16 + e;
                afrag[kt].h[e] = Ah[l15 * NTAGS + K];
            }
        }
        FAcc acc = {};
        #pragma unroll
        for (int kt = 0; kt < 4; ++kt)
            acc.v = __builtin_amdgcn_wmma_f32_16x16x32_f16(
                        false, afrag[kt].v, false, bfrag[kt].v,
                        (short)0, acc.v, false, false);

        // phase 3: alpha_new = rowm[M] + cmax[N] + log(acc) + emission
        const float cm = cmax[Nglob];
        #pragma unroll
        for (int r = 0; r < 8; ++r) {
            int M = hi * 8 + r;   // C/D layout: lanes 0-15 -> M=r, 16-31 -> M=8+r
            float e_t = emissions[((size_t)(bRow0 + M) * SEQ + t) * NTAGS + Nglob];
            alpha[M * NTAGS + Nglob] = rowm[M] + cm + __logf(acc.f[r]) + e_t;
        }
        __syncthreads();
    }

    // --- final logsumexp over tags, per batch row ---
    {
        float av[8], m = -3.0e38f;
        #pragma unroll
        for (int i = 0; i < 8; ++i) {
            av[i] = alpha[myrow * NTAGS + cbase + i];
            m = fmaxf(m, av[i]);
        }
        #pragma unroll
        for (int off = 8; off >= 1; off >>= 1) m = fmaxf(m, __shfl_xor(m, off, 16));
        float s = 0.f;
        #pragma unroll
        for (int i = 0; i < 8; ++i) s += __expf(av[i] - m);
        #pragma unroll
        for (int off = 8; off >= 1; off >>= 1) s += __shfl_xor(s, off, 16);
        if ((tid & 15) == 0) lsered[myrow] = m + __logf(s);
    }
    __syncthreads();
    if (tid == 0) {
        float p = 0.f;
        for (int r = 0; r < MROWS; ++r) p += lsered[r];
        partials[blockIdx.x] = p;
    }
}

// ---------------------------------------------------------------------------
// Gold score: emissions at gold tags + transitions between consecutive tags,
// masked. Deterministic in-block tree reduction -> one partial per block.
// ---------------------------------------------------------------------------
__global__ __launch_bounds__(256) void crf_gold_kernel(
    const float* __restrict__ emissions,
    const long long* __restrict__ tags,
    const unsigned char* __restrict__ mask,
    const float* __restrict__ transitions,
    float* __restrict__ partials)
{
    __shared__ float red[256];
    int idx = blockIdx.x * 256 + threadIdx.x;   // flat (b, t), total B*S
    float v = 0.f;
    if (idx < BATCH * SEQ) {
        int t   = idx & (SEQ - 1);
        int tag = (int)tags[idx];
        float mv = mask[idx] ? 1.f : 0.f;
        float e  = emissions[(size_t)idx * NTAGS + tag];
        float tr = 0.f;
        if (t > 0) {
            int pt = (int)tags[idx - 1];
            tr = transitions[pt * NTAGS + tag];
        }
        v = mv * (e + tr);
    }
    red[threadIdx.x] = v;
    for (int off = 128; off > 0; off >>= 1) {
        __syncthreads();
        if (threadIdx.x < off) red[threadIdx.x] += red[threadIdx.x + off];
    }
    if (threadIdx.x == 0) partials[blockIdx.x] = red[0];
}

// ---------------------------------------------------------------------------
// Deterministic final combine: out = sum(forward partials) - sum(gold partials)
// ---------------------------------------------------------------------------
__global__ void crf_final_kernel(const float* __restrict__ fw,
                                 const float* __restrict__ gold,
                                 float* __restrict__ out)
{
    float p = 0.f;
    for (int i = 0; i < BATCH / MROWS; ++i) p += fw[i];
    float g = 0.f;
    for (int i = 0; i < (BATCH * SEQ) / 256; ++i) g += gold[i];
    out[0] = p - g;
}

extern "C" void kernel_launch(void* const* d_in, const int* in_sizes, int n_in,
                              void* d_out, int out_size, void* d_ws, size_t ws_size,
                              hipStream_t stream) {
    (void)in_sizes; (void)n_in; (void)out_size; (void)ws_size;
    const float*         emissions   = (const float*)d_in[0];
    const long long*     tags        = (const long long*)d_in[1];
    const unsigned char* mask        = (const unsigned char*)d_in[2];
    const float*         transitions = (const float*)d_in[3];
    float* out       = (float*)d_out;
    float* fw_part   = (float*)d_ws;          // 16 floats
    float* gold_part = fw_part + 16;          // 512 floats

    crf_gold_kernel<<<(BATCH * SEQ) / 256, 256, 0, stream>>>(
        emissions, tags, mask, transitions, gold_part);
    crf_forward_kernel<<<BATCH / MROWS, 256, 0, stream>>>(
        emissions, transitions, fw_part);
    crf_final_kernel<<<1, 1, 0, stream>>>(fw_part, gold_part, out);
}
